// MACE_45767171506193
// MI455X (gfx1250) — matrix operand order repaired and checked by reading, not compiled, not run on hardware
//
#include <hip/hip_runtime.h>
#include <hip/hip_bf16.h>

typedef __attribute__((ext_vector_type(16))) _Float16 v16h;
typedef __attribute__((ext_vector_type(8)))  float    v8f;
typedef unsigned int u32x4 __attribute__((ext_vector_type(4)));
typedef int          i32x8 __attribute__((ext_vector_type(8)));
typedef int          i32x4 __attribute__((ext_vector_type(4)));

#define NN      8192
#define EE      131072
#define GG      32
#define FF      64
#define TT      2
#define DD      128
#define NUM_EL  5
#define NBASIS  8
#define SH      9
#define KREAD   576      /* F * SH */
#define H2      192      /* F * 3  */
#define R_MAX   2.03f
#define PI_F    3.14159265358979323846f

__device__ __forceinline__ float softplusf(float x) {
  return (x > 20.f) ? x : log1pf(expf(x));
}
__device__ __forceinline__ float siluf(float x) {
  return x / (1.f + expf(-x));
}
// A-fragment K offset for VGPR index v and lane-half hh (16-bit 16x32 A layout)
__device__ __forceinline__ int a_klo(int v, int hh) {
  return (v < 4) ? (2*v + 8*hh) : (16 + 2*(v-4) + 8*hh);
}
// Low 32 bits of a generic pointer to LDS == LDS byte offset (aperture rule)
__device__ __forceinline__ unsigned lds_off(const void* p) {
  return (unsigned)(unsigned long long)(uintptr_t)p;
}

// ---------------------------------------------------------------------------
// Tensor Data Mover: 2-D f32 tile (tile_rows x tile_cols) from row-major
// tensor (tensor_rows x tensor_cols, row stride row_stride elements) -> LDS.
// D# packing per CDNA5 ISA.
// ---------------------------------------------------------------------------
__device__ __forceinline__ void tdm_load_2d_f32(unsigned lds_byte_off,
                                                const void* gptr,
                                                unsigned tile_rows,
                                                unsigned tile_cols,
                                                unsigned tensor_cols,
                                                unsigned tensor_rows,
                                                unsigned row_stride) {
  unsigned long long ga = (unsigned long long)(uintptr_t)gptr;
  u32x4 g0;
  g0[0] = 1u;                                           // count=1 (valid), user mode
  g0[1] = lds_byte_off;                                 // lds_addr
  g0[2] = (unsigned)(ga & 0xffffffffu);                 // global_addr[31:0]
  g0[3] = (unsigned)((ga >> 32) & 0x01ffffffu) | (2u << 30); // addr[56:32] | type=2
  i32x8 g1;
  g1[0] = (int)(2u << 16);                              // wg_mask=0, data_size=4B
  g1[1] = (int)((tensor_cols & 0xffffu) << 16);         // tensor_dim0[15:0]
  g1[2] = (int)(((tensor_cols >> 16) & 0xffffu) |
                ((tensor_rows & 0xffffu) << 16));       // dim0[31:16] | dim1[15:0]
  g1[3] = (int)(((tensor_rows >> 16) & 0xffffu) |
                ((tile_cols & 0xffffu) << 16));         // dim1[31:16] | tile_dim0
  g1[4] = (int)(tile_rows & 0xffffu);                   // tile_dim1 (tile_dim2=0)
  g1[5] = (int)row_stride;                              // tensor_dim0_stride[31:0]
  g1[6] = 0;                                            // stride0[47:32], stride1 lo
  g1[7] = 0;
  i32x4 z = {0, 0, 0, 0};
#if __clang_major__ >= 23
  i32x8 z8 = {0, 0, 0, 0, 0, 0, 0, 0};
  __builtin_amdgcn_tensor_load_to_lds(g0, g1, z, z, z8, 0);
#else
  __builtin_amdgcn_tensor_load_to_lds(g0, g1, z, z, 0);
#endif
}

// ---------------------------------------------------------------------------
// Per-edge geometry: Y[9], radial basis[8], envelope (pre-scaled by 1/AVG_NEIGH)
// geom layout per edge (18 f32): [Y0..Y8, b0..b7, env/16]
// ---------------------------------------------------------------------------
__global__ void edge_geom_kernel(const float* __restrict__ pos,
                                 const int*   __restrict__ senders,
                                 const int*   __restrict__ receivers,
                                 float*       __restrict__ geom) {
  int e = blockIdx.x * blockDim.x + threadIdx.x;
  if (e >= EE) return;
  int s = senders[e], r = receivers[e];
  float vx = pos[r*3+0] - pos[s*3+0];
  float vy = pos[r*3+1] - pos[s*3+1];
  float vz = pos[r*3+2] - pos[s*3+2];
  float d  = sqrtf(vx*vx + vy*vy + vz*vz);
  float ds = fmaxf(d, 1e-9f);
  float inv = 1.f / ds;
  float x = vx*inv, y = vy*inv, z = vz*inv;
  const float c1 = 1.7320508075688772f;   // sqrt(3)
  const float c2 = 3.8729833462074170f;   // sqrt(15)
  const float c3 = 1.1180339887498949f;   // sqrt(5)/2
  float* g = geom + (size_t)e * 18;
  g[0] = 1.f;
  g[1] = c1 * x;  g[2] = c1 * y;  g[3] = c1 * z;
  g[4] = c2 * x * y;  g[5] = c2 * y * z;
  g[6] = c3 * (3.f * z * z - 1.f);
  g[7] = c2 * x * z;
  g[8] = 0.5f * c2 * (x*x - y*y);
  const float bc = sqrtf(2.f / R_MAX);
  #pragma unroll
  for (int j = 0; j < NBASIS; ++j)
    g[9 + j] = bc * sinf((float)(j + 1) * PI_F * ds / R_MAX) * inv;
  float tt  = fminf(d / R_MAX, 1.f - 1e-6f);
  float env = (d < R_MAX) ? expf(-tt * tt / (1.f - tt * tt)) : 0.f;
  g[17] = env * (1.f / 16.f);  // fold 1/AVG_NEIGH into envelope
}

// ---------------------------------------------------------------------------
// x[n][f] = emb_table[species[n]][f]
// ---------------------------------------------------------------------------
__global__ void node_init_kernel(const int* __restrict__ species,
                                 const float* __restrict__ emb_table,
                                 float* __restrict__ x) {
  int i = blockIdx.x * blockDim.x + threadIdx.x;
  if (i >= NN * FF) return;
  int n = i >> 6, f = i & 63;
  x[i] = emb_table[species[n] * FF + f];
}

// ---------------------------------------------------------------------------
// Fused edge kernel (per layer):
//   h = softplus(basis @ W1) @ W2 * env   (WMMA f16, 16 edges per wave)
//   msg scatter: agg[recv][f*9+m] += x[send][f] * Y[m] * h[f*3+L_IDX[m]]
// Block = 256 threads = 8 waves = 128 edges. W2 staged to LDS by the TDM.
// ---------------------------------------------------------------------------
__global__ void __launch_bounds__(256)
edge_msg_kernel(const float* __restrict__ geom,
                const int*   __restrict__ senders,
                const int*   __restrict__ receivers,
                const float* __restrict__ x,
                const float* __restrict__ W1,   // 8 x 64   (layer slice)
                const float* __restrict__ W2,   // 64 x 192 (layer slice)
                float*       __restrict__ agg) {
  __shared__ float W2s[FF * H2];     // 48 KB, raw f32 via TDM
  __shared__ float W1s[NBASIS * FF];
  __shared__ float Ys[8][16][SH];
  __shared__ float Es[8][16];
  __shared__ int   Snd[8][16];
  __shared__ int   Rcv[8][16];

  const int t    = threadIdx.x;
  const int lane = t & 31, wv = t >> 5;
  const int hh   = lane >> 4, mrow = lane & 15;
  const int ncol = lane & 15;

  // ---- DMA W2 (64 x 192 f32) into LDS with the Tensor Data Mover ----
  if (wv == 0)
    tdm_load_2d_f32(lds_off(W2s), W2, /*tile_rows=*/FF, /*tile_cols=*/H2,
                    /*tensor_cols=*/H2, /*tensor_rows=*/FF, /*row_stride=*/H2);

  for (int i = t; i < NBASIS * FF; i += 256) W1s[i] = W1[i];

  const int e = blockIdx.x * 128 + wv * 16 + mrow;
  if (lane < 16) {
    const float* g = geom + (size_t)e * 18;
    #pragma unroll
    for (int j = 0; j < SH; ++j) Ys[wv][mrow][j] = g[j];
    Es[wv][mrow]  = g[17];
    Snd[wv][mrow] = senders[e];
    Rcv[wv][mrow] = receivers[e];
  }
  if (wv == 0) __builtin_amdgcn_s_wait_tensorcnt(0);
  __syncthreads();

  // ---- t16x64 = softplus(basis @ W1), built directly as two A fragments ----
  float bas[NBASIS];
  {
    const float* g = geom + (size_t)e * 18 + 9;
    #pragma unroll
    for (int j = 0; j < NBASIS; ++j) bas[j] = g[j];
  }
  v16h a0, a1;
  #pragma unroll
  for (int v = 0; v < 8; ++v) {
    int klo = a_klo(v, hh);
    #pragma unroll
    for (int p = 0; p < 2; ++p) {
      int k0 = klo + p;
      float s0 = 0.f, s1 = 0.f;
      #pragma unroll
      for (int j = 0; j < NBASIS; ++j) {
        s0 = fmaf(bas[j], W1s[j * FF + k0],      s0);
        s1 = fmaf(bas[j], W1s[j * FF + 32 + k0], s1);
      }
      a0[2*v + p] = (_Float16)softplusf(s0);
      a1[2*v + p] = (_Float16)softplusf(s1);
    }
  }

  // ---- 16x64 @ 64x192 via WMMA, one 16x16 output tile at a time ----
  for (int nt = 0; nt < 12; ++nt) {
    v16h b0, b1;
    #pragma unroll
    for (int v = 0; v < 8; ++v) {
      #pragma unroll
      for (int p = 0; p < 2; ++p) {
        int kl = a_klo(v, hh) + p;
        b0[2*v + p] = (_Float16)W2s[kl        * H2 + nt * 16 + ncol];
        b1[2*v + p] = (_Float16)W2s[(32 + kl) * H2 + nt * 16 + ncol];
      }
    }
    v8f c = {};
    c = __builtin_amdgcn_wmma_f32_16x16x32_f16(false, a0, false, b0, (short)0, c, false, false);
    c = __builtin_amdgcn_wmma_f32_16x16x32_f16(false, a1, false, b1, (short)0, c, false, false);

    // ---- scatter messages: fully unrolled, exec-predicated atomics ----
    const int col  = nt * 16 + ncol;          // h column 0..191
    const int f    = col / 3;
    const int cidx = col - 3 * f;             // L_IDX class of this column
    #pragma unroll
    for (int v = 0; v < 8; ++v) {
      int mm = v + 8 * hh;                    // edge row in tile
      float hval = c[v] * Es[wv][mm];
      int snd = Snd[wv][mm], rcv = Rcv[wv][mm];
      float xh = x[snd * FF + f] * hval;
      float* ap = agg + (size_t)rcv * KREAD + f * SH;
      const float* yrow = Ys[wv][mm];
      // L_IDX = {0,1,1,1,2,2,2,2,2}; static offsets, predicate per component
      #pragma unroll
      for (int msh = 0; msh < SH; ++msh) {
        const int lidx = (msh == 0) ? 0 : ((msh < 4) ? 1 : 2);
        if (lidx == cidx) atomicAdd(ap + msh, xh * yrow[msh]);
      }
    }
  }
}

// ---------------------------------------------------------------------------
// Node kernel (per layer):
//   reads = agg(N x 576) @ W_read(576 x 128)        -> emb[:, li*128 : ]
//   x     = silu(agg[:,:,0] @ W_upd + x @ W_self)   (fused dual-WMMA)
// Block = 256 threads = 8 waves = 128 nodes; each wave owns 16 node rows.
// W_read K-slices are double-buffered via TDM so DMA overlaps WMMA.
// ---------------------------------------------------------------------------
__global__ void __launch_bounds__(256)
node_update_kernel(const float* __restrict__ agg,
                   const float* __restrict__ W_read,  // 576 x 128 layer slice
                   const float* __restrict__ W_upd,   // 64 x 64
                   const float* __restrict__ W_self,  // 64 x 64
                   float*       __restrict__ x,       // N x 64 (in/out)
                   float*       __restrict__ emb,     // N x 256
                   int li) {
  __shared__ float    Ws[2][32 * DD];  // 2 x 16 KB, raw f32 via TDM
  __shared__ _Float16 Wus[FF * FF];
  __shared__ _Float16 Wss[FF * FF];

  const int t    = threadIdx.x;
  const int lane = t & 31, wv = t >> 5;
  const int hh   = lane >> 4, mrow = lane & 15;
  const int ncol = lane & 15;
  const int n0   = blockIdx.x * 128 + wv * 16;

  // prologue: DMA slice 0; stage the small square weights as f16
  if (wv == 0)
    tdm_load_2d_f32(lds_off(Ws[0]), W_read, 32, DD, DD, KREAD, DD);
  for (int i = t; i < FF * FF; i += 256) {
    Wus[i] = (_Float16)W_upd[i];
    Wss[i] = (_Float16)W_self[i];
  }

  v8f c[8] = {};
  for (int ks = 0; ks < 18; ++ks) {          // K = 576 = 18 * 32
    int kb = ks * 32;
    __syncthreads();                         // everyone done with buf[(ks+1)&1]
    if (wv == 0) {
      __builtin_amdgcn_s_wait_tensorcnt(0);  // slice ks has landed in LDS
      if (ks + 1 < 18)                       // prefetch next slice (overlaps WMMA)
        tdm_load_2d_f32(lds_off(Ws[(ks + 1) & 1]),
                        W_read + (size_t)(kb + 32) * DD, 32, DD, DD, KREAD, DD);
    }
    __syncthreads();
    const float* Wb = Ws[ks & 1];
    v16h a;
    #pragma unroll
    for (int v = 0; v < 8; ++v) {
      int klo = a_klo(v, hh);
      #pragma unroll
      for (int p = 0; p < 2; ++p)
        a[2*v + p] = (_Float16)agg[(size_t)(n0 + mrow) * KREAD + kb + klo + p];
    }
    #pragma unroll
    for (int nt = 0; nt < 8; ++nt) {
      v16h b;
      #pragma unroll
      for (int v = 0; v < 8; ++v)
        #pragma unroll
        for (int p = 0; p < 2; ++p)
          b[2*v + p] = (_Float16)Wb[(a_klo(v, hh) + p) * DD + nt * 16 + ncol];
      c[nt] = __builtin_amdgcn_wmma_f32_16x16x32_f16(false, a, false, b, (short)0, c[nt], false, false);
    }
  }
  // write readout into emb
  #pragma unroll
  for (int nt = 0; nt < 8; ++nt)
    #pragma unroll
    for (int v = 0; v < 8; ++v)
      emb[(size_t)(n0 + v + 8*hh) * (TT * DD) + li * DD + nt * 16 + ncol] = c[nt][v];

  // ---- x update: silu(agg[:,:,0] @ W_upd + x @ W_self), K=64, N=64 ----
  v8f c2[4] = {};
  #pragma unroll
  for (int ks = 0; ks < 2; ++ks) {
    int kb = ks * 32;
    v16h au, as;
    #pragma unroll
    for (int v = 0; v < 8; ++v) {
      int klo = a_klo(v, hh);
      #pragma unroll
      for (int p = 0; p < 2; ++p) {
        int k = kb + klo + p;
        au[2*v + p] = (_Float16)agg[(size_t)(n0 + mrow) * KREAD + k * SH]; // m=0 comp
        as[2*v + p] = (_Float16)x[(size_t)(n0 + mrow) * FF + k];
      }
    }
    #pragma unroll
    for (int nt = 0; nt < 4; ++nt) {
      v16h bu, bs;
      #pragma unroll
      for (int v = 0; v < 8; ++v)
        #pragma unroll
        for (int p = 0; p < 2; ++p) {
          int k = kb + a_klo(v, hh) + p;
          bu[2*v + p] = Wus[k * FF + nt * 16 + ncol];
          bs[2*v + p] = Wss[k * FF + nt * 16 + ncol];
        }
      c2[nt] = __builtin_amdgcn_wmma_f32_16x16x32_f16(false, au, false, bu, (short)0, c2[nt], false, false);
      c2[nt] = __builtin_amdgcn_wmma_f32_16x16x32_f16(false, as, false, bs, (short)0, c2[nt], false, false);
    }
  }
  // all reads of x for this wave's rows are done; safe to overwrite in place
  #pragma unroll
  for (int nt = 0; nt < 4; ++nt)
    #pragma unroll
    for (int v = 0; v < 8; ++v)
      x[(size_t)(n0 + v + 8*hh) * FF + nt * 16 + ncol] = siluf(c2[nt][v]);
}

// ---------------------------------------------------------------------------
// focus_logits[n] = emb[n] . w_focus
// ---------------------------------------------------------------------------
__global__ void focus_kernel(const float* __restrict__ emb,
                             const float* __restrict__ w_focus,
                             float* __restrict__ out) {
  int n = blockIdx.x * blockDim.x + threadIdx.x;
  if (n >= NN) return;
  const float* e = emb + (size_t)n * (TT * DD);
  float s = 0.f;
  #pragma unroll 8
  for (int i = 0; i < TT * DD; ++i) s = fmaf(e[i], w_focus[i], s);
  out[n] = s;
}

// ---------------------------------------------------------------------------
// Per-graph heads: species_logits = softplus(femb@Ws1)@Ws2,
//                  pos = (species_table[ts]*femb) @ W_pos
// One block per graph.
// ---------------------------------------------------------------------------
__global__ void __launch_bounds__(256)
graph_head_kernel(const float* __restrict__ emb,
                  const int*   __restrict__ n_node,
                  const int*   __restrict__ target_species,
                  const float* __restrict__ Ws1,            // 256 x 128
                  const float* __restrict__ Ws2,            // 128 x 5
                  const float* __restrict__ species_table,  // 5 x 256
                  const float* __restrict__ W_pos,          // 256 x 1024
                  float* __restrict__ out_species,
                  float* __restrict__ out_pos) {
  __shared__ float fe[TT * DD];
  __shared__ float sf[TT * DD];
  __shared__ float hid[128];
  int g = blockIdx.x, t = threadIdx.x;
  int first = 0;
  for (int i = 0; i < g; ++i) first += n_node[i];
  fe[t] = emb[(size_t)first * (TT * DD) + t];
  int ts = target_species[g];
  sf[t] = species_table[ts * (TT * DD) + t] * fe[t];
  __syncthreads();
  if (t < 128) {
    float s = 0.f;
    for (int i = 0; i < TT * DD; ++i) s = fmaf(fe[i], Ws1[i * 128 + t], s);
    hid[t] = softplusf(s);
  }
  __syncthreads();
  if (t < NUM_EL) {
    float s = 0.f;
    for (int j = 0; j < 128; ++j) s = fmaf(hid[j], Ws2[j * NUM_EL + t], s);
    out_species[g * NUM_EL + t] = s;
  }
  for (int o = t; o < 1024; o += 256) {
    float s = 0.f;
    for (int i = 0; i < TT * DD; ++i) s = fmaf(sf[i], W_pos[i * 1024 + o], s);
    out_pos[(size_t)g * 1024 + o] = s;
  }
}

// ---------------------------------------------------------------------------
extern "C" void kernel_launch(void* const* d_in, const int* in_sizes, int n_in,
                              void* d_out, int out_size, void* d_ws, size_t ws_size,
                              hipStream_t stream) {
  (void)in_sizes; (void)n_in; (void)out_size; (void)ws_size;
  const float* positions      = (const float*)d_in[0];
  const int*   species        = (const int*)  d_in[1];
  const int*   senders        = (const int*)  d_in[2];
  const int*   receivers      = (const int*)  d_in[3];
  const int*   n_node         = (const int*)  d_in[4];
  const int*   target_species = (const int*)  d_in[5];
  const float* emb_table      = (const float*)d_in[6];
  const float* W1             = (const float*)d_in[7];
  const float* W2             = (const float*)d_in[8];
  const float* W_self         = (const float*)d_in[9];
  const float* W_upd          = (const float*)d_in[10];
  const float* W_read         = (const float*)d_in[11];
  const float* w_focus        = (const float*)d_in[12];
  const float* Ws1            = (const float*)d_in[13];
  const float* Ws2            = (const float*)d_in[14];
  const float* species_table  = (const float*)d_in[15];
  const float* W_pos          = (const float*)d_in[16];

  char* ws = (char*)d_ws;
  size_t off = 0;
  float* geom = (float*)(ws + off); off += (size_t)EE * 18 * sizeof(float);     // 9.4 MB
  float* x    = (float*)(ws + off); off += (size_t)NN * FF * sizeof(float);     // 2.0 MB
  float* agg  = (float*)(ws + off); off += (size_t)NN * KREAD * sizeof(float);  // 18.9 MB
  float* emb  = (float*)(ws + off);                                             // 8.4 MB

  float* out_focus   = (float*)d_out;
  float* out_species = out_focus + NN;
  float* out_pos     = out_species + GG * NUM_EL;

  edge_geom_kernel<<<EE / 256, 256, 0, stream>>>(positions, senders, receivers, geom);
  node_init_kernel<<<(NN * FF) / 256, 256, 0, stream>>>(species, emb_table, x);

  for (int li = 0; li < TT; ++li) {
    (void)hipMemsetAsync(agg, 0, (size_t)NN * KREAD * sizeof(float), stream);
    edge_msg_kernel<<<EE / 128, 256, 0, stream>>>(
        geom, senders, receivers, x,
        W1 + (size_t)li * NBASIS * FF,
        W2 + (size_t)li * FF * H2, agg);
    node_update_kernel<<<NN / 128, 256, 0, stream>>>(
        agg,
        W_read + (size_t)li * KREAD * DD,
        W_upd + (size_t)li * FF * FF,
        W_self + (size_t)li * FF * FF,
        x, emb, li);
  }

  focus_kernel<<<NN / 256, 256, 0, stream>>>(emb, w_focus, out_focus);
  graph_head_kernel<<<GG, 256, 0, stream>>>(emb, n_node, target_species,
                                            Ws1, Ws2, species_table, W_pos,
                                            out_species, out_pos);
}
